// PostProcess_11819749999250
// MI455X (gfx1250) — matrix operand order repaired and checked by reading, not compiled, not run on hardware
//
#include <hip/hip_runtime.h>
#include <hip/hip_bf16.h>

// ---------------------------------------------------------------------------
// Post-process pipeline: hysteresis (26-conn CCL) -> z-closing (fused WMMA
// banded matmuls, dilate+erode in one kernel) -> small-object removal
// (6-conn CCL + size filter).
// Shapes fixed by the reference: (B=2, D=24, H=448, W=448) float32 in [0,1).
// ---------------------------------------------------------------------------

#define BB 2
#define DD 24
#define HH 448
#define WW 448
#define HW (HH * WW)            // 200704
#define NTOT (BB * DD * HW)     // 9633792

#define T_LOW 0.5f
#define T_HIGH 0.9f
#define MIN_SIZE 100

typedef __attribute__((ext_vector_type(16))) _Float16 v16h;
typedef __attribute__((ext_vector_type(8)))  float    v8f;

// ---------------------------- thresholding ---------------------------------
__global__ void threshold_kernel(const float* __restrict__ in,
                                 unsigned char* __restrict__ weak,
                                 unsigned char* __restrict__ strong) {
    int i = blockIdx.x * blockDim.x + threadIdx.x;
    if (i >= NTOT) return;
    float v = in[i];
    weak[i]   = (v >= T_LOW)  ? 1 : 0;
    strong[i] = (v >= T_HIGH) ? 1 : 0;
}

// ---------------------------- union-find CCL -------------------------------
__device__ __forceinline__ int find_root(const int* __restrict__ L, int x) {
    int p = L[x];
    while (p != x) { x = p; p = L[x]; }
    return x;
}

// Komura-style union via atomicMin root chasing; parent < child invariant
// guarantees termination; a single merge pass reaches the global fixed point.
__device__ __forceinline__ void unite(int* L, int a, int b) {
    int ra = find_root(L, a);
    int rb = find_root(L, b);
    while (ra != rb) {
        int hi = ra > rb ? ra : rb;
        int lo = ra > rb ? rb : ra;
        int prev = atomicMin(&L[hi], lo);
        if (prev == hi) break;          // hi was a root; now attached to lo
        ra = find_root(L, prev);        // keep merging displaced parent
        rb = find_root(L, lo);
    }
}

__global__ void ccl_init_kernel(const unsigned char* __restrict__ mask,
                                int* __restrict__ L) {
    int i = blockIdx.x * blockDim.x + threadIdx.x;
    if (i >= NTOT) return;
    L[i] = mask[i] ? i : -1;
}

// 26-connectivity merge (13 lexicographically-negative neighbors); z never
// crosses the batch boundary because we decode (b, z) explicitly.
__global__ void ccl_merge26_kernel(const unsigned char* __restrict__ mask,
                                   int* __restrict__ L) {
    int i = blockIdx.x * blockDim.x + threadIdx.x;
    if (i >= NTOT) return;
    if (!mask[i]) return;
    int x = i % WW;
    int y = (i / WW) % HH;
    int z = (i / HW) % DD;

    if (z > 0) {
        #pragma unroll
        for (int dy = -1; dy <= 1; ++dy) {
            if (y + dy < 0 || y + dy >= HH) continue;
            #pragma unroll
            for (int dx = -1; dx <= 1; ++dx) {
                if (x + dx < 0 || x + dx >= WW) continue;
                int j = i - HW + dy * WW + dx;
                if (mask[j]) unite(L, i, j);
            }
        }
    }
    if (y > 0) {
        #pragma unroll
        for (int dx = -1; dx <= 1; ++dx) {
            if (x + dx < 0 || x + dx >= WW) continue;
            int j = i - WW + dx;
            if (mask[j]) unite(L, i, j);
        }
    }
    if (x > 0) {
        int j = i - 1;
        if (mask[j]) unite(L, i, j);
    }
}

// 6-connectivity merge (-z, -y, -x)
__global__ void ccl_merge6_kernel(const unsigned char* __restrict__ mask,
                                  int* __restrict__ L) {
    int i = blockIdx.x * blockDim.x + threadIdx.x;
    if (i >= NTOT) return;
    if (!mask[i]) return;
    int x = i % WW;
    int y = (i / WW) % HH;
    int z = (i / HW) % DD;
    if (z > 0 && mask[i - HW]) unite(L, i, i - HW);
    if (y > 0 && mask[i - WW]) unite(L, i, i - WW);
    if (x > 0 && mask[i - 1])  unite(L, i, i - 1);
}

__global__ void ccl_flatten_kernel(const unsigned char* __restrict__ mask,
                                   int* __restrict__ L) {
    int i = blockIdx.x * blockDim.x + threadIdx.x;
    if (i >= NTOT) return;
    if (!mask[i]) return;
    L[i] = find_root(L, i);
}

// ---------------------------- hysteresis via CCL ---------------------------
__global__ void zero_i32_kernel(int* __restrict__ p) {
    int i = blockIdx.x * blockDim.x + threadIdx.x;
    if (i >= NTOT) return;
    p[i] = 0;
}

__global__ void hyst_flag_kernel(const unsigned char* __restrict__ strong,
                                 const int* __restrict__ L,
                                 int* __restrict__ flags) {
    int i = blockIdx.x * blockDim.x + threadIdx.x;
    if (i >= NTOT) return;
    if (strong[i]) flags[L[i]] = 1;   // idempotent store; no atomic needed
}

__global__ void hyst_mask_kernel(const unsigned char* __restrict__ weak,
                                 const int* __restrict__ L,
                                 const int* __restrict__ flags,
                                 unsigned char* __restrict__ mask) {
    int i = blockIdx.x * blockDim.x + threadIdx.x;
    if (i >= NTOT) return;
    mask[i] = (weak[i] && flags[L[i]]) ? 1 : 0;
}

// ---------------- fused z-closing as two WMMA banded matmuls ---------------
// closing(m) = erode(dilate(m)) along z with a (3,1,1) line, zero borders.
// With T = tridiagonal-ones (zero-padded 32x32):
//   dil = (T*m  > 0.5)  (binary OR of existing neighbors)
//   ero = (T*dil > 2.5) (needs all 3 window entries == 1; pad rows can never
//                        reach 3 -> scipy's forced-zero z borders fall out)
// One wave owns a 16-column tile and ALL 24 z values, so both matmuls fuse in
// registers. Between them we convert the D-matrix accumulator layout
// (row M = h*8+r on lane-half h) to the B-matrix operand layout
// (K = h*16+e) with one __shfl_xor(...,16) partner-half exchange.
// Grid is exact (every wave full) -> EXEC all ones at WMMA, as required.
__global__ void closez_wmma_kernel(const unsigned char* __restrict__ src,
                                   unsigned char* __restrict__ dst) {
    int wave = (blockIdx.x * blockDim.x + threadIdx.x) >> 5;
    int lane = threadIdx.x & 31;
    int n = lane & 15;       // column within tile / row index for A
    int h = lane >> 4;       // lane-half selects K ranges per ISA layout

    int col  = wave * 16 + n;            // flattened (b, y, x)
    int b    = col / HW;
    int rest = col % HW;
    int base = b * DD * HW + rest;       // voxel index of (b, z=0, y, x)

    // B operand 32x16 f16: lane-half h holds K = h*16 + e (e = 0..15).
    v16h bvec;
    #pragma unroll
    for (int e = 0; e < 16; ++e) {
        int K = h * 16 + e;
        float v = (K < DD) ? (float)src[base + K * HW] : 0.0f;
        bvec[e] = (_Float16)v;
    }

    // A operand tiles 16x32 f16 (constant tridiagonal from lane id):
    // vector element e -> K(e) = h*16 + e.
    v16h atop, abot;
    #pragma unroll
    for (int e = 0; e < 16; ++e) {
        int K  = h * 16 + e;
        int mt = n;
        int mb = 16 + n;
        atop[e] = (_Float16)(((K - mt) <= 1 && (mt - K) <= 1) ? 1.0f : 0.0f);
        abot[e] = (_Float16)(((K - mb) <= 1 && (mb - K) <= 1) ? 1.0f : 0.0f);
    }

    v8f c0 = {};

    // ---- pass 1: dilation sums ----
    v8f dtop = __builtin_amdgcn_wmma_f32_16x16x32_f16(
        false, atop, false, bvec, (short)0, c0, false, false);
    v8f dbot = __builtin_amdgcn_wmma_f32_16x16x32_f16(
        false, abot, false, bvec, (short)0, c0, false, false);

    // D layout -> B layout: dil[M] lives at dtop[M-h*8] / dbot[M-16-h*8] on
    // lane-half h. Each lane needs the partner half's 8 rows.
    float ttop[8], tbot[8];
    #pragma unroll
    for (int r = 0; r < 8; ++r) {
        ttop[r] = __shfl_xor(dtop[r], 16, 32);   // partner half's dtop rows
        tbot[r] = __shfl_xor(dbot[r], 16, 32);   // partner half's dbot rows
    }
    v16h dil;   // B operand for erosion: K = h*16 + e, zero for K >= DD
    #pragma unroll
    for (int e = 0; e < 16; ++e) {
        float s;
        if (h == 0) {
            // K = e: rows 0..7 from own dtop, rows 8..15 from partner dtop
            s = (e < 8) ? dtop[e] : ttop[e - 8];
        } else {
            // K = 16+e: rows 16..23 from partner (h=0) dbot; K >= 24 is pad
            s = (e < 8) ? tbot[e] : -1.0f;       // -1 -> binarizes to 0
        }
        dil[e] = (_Float16)((s > 0.5f) ? 1.0f : 0.0f);
    }

    // ---- pass 2: erosion sums ----
    v8f etop = __builtin_amdgcn_wmma_f32_16x16x32_f16(
        false, atop, false, dil, (short)0, c0, false, false);
    v8f ebot = __builtin_amdgcn_wmma_f32_16x16x32_f16(
        false, abot, false, dil, (short)0, c0, false, false);

    // store: accumulator VGPR r holds row M = h*8 + r (top) / 16+h*8+r (bot)
    #pragma unroll
    for (int r = 0; r < 8; ++r) {
        int mt = h * 8 + r;                       // 0..15, always < DD
        dst[base + mt * HW] = (etop[r] > 2.5f) ? 1 : 0;
        int mb = 16 + h * 8 + r;                  // 16..31
        if (mb < DD) dst[base + mb * HW] = (ebot[r] > 2.5f) ? 1 : 0;
    }
}

// ---------------------------- size filter ----------------------------------
__global__ void count_kernel(const unsigned char* __restrict__ mask,
                             const int* __restrict__ L,
                             int* __restrict__ counts) {
    int i = blockIdx.x * blockDim.x + threadIdx.x;
    if (i >= NTOT) return;
    if (mask[i]) atomicAdd(&counts[L[i]], 1);
}

__global__ void final_kernel(const unsigned char* __restrict__ mask,
                             const int* __restrict__ L,
                             const int* __restrict__ counts,
                             float* __restrict__ out) {
    int i = blockIdx.x * blockDim.x + threadIdx.x;
    if (i >= NTOT) return;
    out[i] = (mask[i] && counts[L[i]] >= MIN_SIZE) ? 1.0f : 0.0f;
}

// ---------------------------------------------------------------------------
extern "C" void kernel_launch(void* const* d_in, const int* in_sizes, int n_in,
                              void* d_out, int out_size, void* d_ws, size_t ws_size,
                              hipStream_t stream) {
    const float* outputs = (const float*)d_in[0];
    float* out = (float*)d_out;

    // Workspace layout (12*N bytes total ~= 116 MB):
    unsigned char* weak   = (unsigned char*)d_ws;          // N
    unsigned char* strong = weak + NTOT;                   // N
    unsigned char* mask   = strong + NTOT;                 // N (hysteresis)
    unsigned char* mask2  = mask + NTOT;                   // N (after closing)
    int* L      = (int*)(mask2 + NTOT);                    // 4N
    int* counts = L + NTOT;                                // 4N (flags/counts)

    const int TPB = 256;
    const int NB  = (NTOT + TPB - 1) / TPB;                // 37632 exact

    // 1) thresholds
    threshold_kernel<<<NB, TPB, 0, stream>>>(outputs, weak, strong);

    // 2) hysteresis == 26-conn components of `weak` that touch `strong`
    ccl_init_kernel<<<NB, TPB, 0, stream>>>(weak, L);
    ccl_merge26_kernel<<<NB, TPB, 0, stream>>>(weak, L);
    ccl_flatten_kernel<<<NB, TPB, 0, stream>>>(weak, L);
    zero_i32_kernel<<<NB, TPB, 0, stream>>>(counts);
    hyst_flag_kernel<<<NB, TPB, 0, stream>>>(strong, L, counts);
    hyst_mask_kernel<<<NB, TPB, 0, stream>>>(weak, L, counts, mask);

    // 3) fused binary closing along z (dilate + erode, 4 WMMAs per wave)
    //    2*448*448/16 = 25088 tiles, 8 waves/block -> 3136 blocks exact
    const int numTiles = (BB * HW) / 16;
    const int wmmaBlocks = (numTiles * 32) / TPB;
    closez_wmma_kernel<<<wmmaBlocks, TPB, 0, stream>>>(mask, mask2);

    // 4) remove small objects: 6-conn CCL + size histogram + filter
    ccl_init_kernel<<<NB, TPB, 0, stream>>>(mask2, L);
    ccl_merge6_kernel<<<NB, TPB, 0, stream>>>(mask2, L);
    ccl_flatten_kernel<<<NB, TPB, 0, stream>>>(mask2, L);
    zero_i32_kernel<<<NB, TPB, 0, stream>>>(counts);
    count_kernel<<<NB, TPB, 0, stream>>>(mask2, L, counts);
    final_kernel<<<NB, TPB, 0, stream>>>(mask2, L, counts, out);
}
